// Net_66133906424401
// MI455X (gfx1250) — compile-verified, matrix operand order checked by reading, not run on hardware
//
#include <hip/hip_runtime.h>
#include <hip/hip_bf16.h>
#include <stdint.h>

// ---------------------------------------------------------------------------
// 10-layer stacked LSTM (B=1024, H=1024) + 3 dense layers on gfx1250.
// gates = X @ Wih^T + H0 @ Whh^T + biases.
//  - fp32 accumulation via v_wmma_f32_16x16x32_bf16 (wave32)
//  - activations carried as bf16; A-tiles DMA'd with global_load_async_to_lds
//  - fp32 operands (weights, h0): batched b128 loads -> cvt_pk_bf16 -> ds_store
//  - K-loop software-pipelined with double-buffered LDS (1 barrier / K-step)
//  - K=17 inputs pre-padded to K=32 so all GEMM loops are unguarded
// ---------------------------------------------------------------------------

typedef __attribute__((ext_vector_type(16))) __bf16 v16bf;
typedef __attribute__((ext_vector_type(8)))  __bf16 v8bf;
typedef __attribute__((ext_vector_type(4)))  __bf16 v4bf;
typedef __attribute__((ext_vector_type(8)))  float  v8f;
typedef __attribute__((ext_vector_type(4)))  float  v4f;

#define BM        128   // batch rows per workgroup
#define BN        32    // output columns per workgroup
#define LSTR      40    // LDS row stride (bf16): 80B -> conflict-free b128 pattern
#define NTHREADS  256   // 8 wave32
#define HDIM      1024

__device__ __forceinline__ float fsig(float x) {
  return 1.0f / (1.0f + __expf(-x));
}
__device__ __forceinline__ float ftanh_fast(float x) {
  float e = __expf(2.0f * x);
  return (e - 1.0f) / (e + 1.0f);
}

// 16x32 bf16 fragment (A layout; B symmetric with lane->column).
// Lane L holds row (L&15); K-chunks at {c0, c0+16}, c0 = (L>>4)*8.
__device__ __forceinline__ v16bf load_frag(const __bf16* base, int row, int lane) {
  const int c0 = (lane >> 4) << 3;
  const __bf16* p = base + row * LSTR + c0;
  v8bf lo = *(const v8bf*)(p);
  v8bf hi = *(const v8bf*)(p + 16);
  return __builtin_shufflevector(lo, hi, 0, 1, 2, 3, 4, 5, 6, 7,
                                         8, 9, 10, 11, 12, 13, 14, 15);
}

__device__ __forceinline__ v4bf cvt4(v4f w) {
  v4bf o;
  o[0] = (__bf16)w[0]; o[1] = (__bf16)w[1];
  o[2] = (__bf16)w[2]; o[3] = (__bf16)w[3];
  return o;
}

// acc[g][ct] += X(BM x Klim) * W(rows g*gstride+n0.., K)^T.
// ABF: A operand is bf16 in global -> async DMA straight to LDS.
// Requires Klim % 32 == 0 (callers pre-pad). sA/sW are double buffers.
template <int NG, bool ABF>
__device__ __forceinline__ void mm_accum(
    const void* __restrict__ Xv, int ldx, int Klim,
    const float* __restrict__ W, int ldw, int gstride, int Nout,
    int m0, int n0, int tid,
    __bf16* sA, __bf16* sW, v8f acc[NG][2])
{
  const int lane  = tid & 31;
  const int rbase = (tid >> 5) << 4;        // 16-row stripe per wave
  // async A staging: 2 threads per row, 32B halves
  const int ar = tid >> 1;
  const int ac = (tid & 1) << 4;
  // fp32 staging coords (thread-invariant across the 4 unrolled chunks)
  const int fr = tid >> 3;                  // row base (0..31), rows fr+32j
  const int fc = (tid & 7) << 2;            // 4-float column group

  __bf16* sA1 = sA + BM * LSTR;
  __bf16* sW1 = sW + NG * BN * LSTR;

  const size_t gofs   = (size_t)gstride * ldw;         // gate stride in W
  const float* wbase  = W + (size_t)(n0 + fr) * ldw + fc;
  const bool   wvalid = (n0 + fr) < Nout;

  v4f wreg[NG];
  v4f areg[4];

  auto loadW = [&](int kk) {
#pragma unroll
    for (int g = 0; g < NG; ++g) {
      v4f w = {0.f, 0.f, 0.f, 0.f};
      const float* p = wbase + (size_t)g * gofs + kk;
      if (wvalid) {
        w = *(const v4f*)p;
        if (kk + 32 < Klim) __builtin_prefetch(p + 32, 0, 1);  // global_prefetch_b8
      }
      wreg[g] = w;
    }
  };
  auto storeW = [&](__bf16* sWb) {
#pragma unroll
    for (int g = 0; g < NG; ++g)
      *(v4bf*)&sWb[(g * BN + fr) * LSTR + fc] = cvt4(wreg[g]);
  };
  auto loadA = [&](int kk) {
    const float* p = (const float*)Xv + (size_t)(m0 + fr) * ldx + kk + fc;
    const size_t rstep = (size_t)ldx * 32;
#pragma unroll
    for (int j = 0; j < 4; ++j) areg[j] = *(const v4f*)(p + j * rstep);
  };
  auto storeA = [&](__bf16* sAb) {
#pragma unroll
    for (int j = 0; j < 4; ++j)
      *(v4bf*)&sAb[(fr + 32 * j) * LSTR + fc] = cvt4(areg[j]);
  };
  auto asyncA = [&](int kk, __bf16* sAb) {
    const __bf16* gp = (const __bf16*)Xv + (size_t)(m0 + ar) * ldx + kk + ac;
    const unsigned lds0 = (unsigned)(uintptr_t)(&sAb[ar * LSTR + ac]);
    const unsigned long long ga = (unsigned long long)(uintptr_t)gp;
    // inst offset is added to BOTH the LDS and the global address (ISA ch.10)
    asm volatile("global_load_async_to_lds_b128 %0, %1, off"
                 :: "v"(lds0), "v"(ga) : "memory");
    asm volatile("global_load_async_to_lds_b128 %0, %1, off offset:16"
                 :: "v"(lds0), "v"(ga) : "memory");
  };
  auto compute = [&](const __bf16* cA, const __bf16* cW) {
    const v16bf a = load_frag(cA + rbase * LSTR, lane & 15, lane);
#pragma unroll
    for (int g = 0; g < NG; ++g) {
#pragma unroll
      for (int ct = 0; ct < 2; ++ct) {
        const v16bf b = load_frag(cW + (g * BN + ct * 16) * LSTR, lane & 15, lane);
        acc[g][ct] = __builtin_amdgcn_wmma_f32_16x16x32_bf16(
            false, a, false, b, (short)0, acc[g][ct], false, false);
      }
    }
  };

  // ---- prologue: stage tile 0 into buffer 0 --------------------------------
  loadW(0);
  if constexpr (ABF) asyncA(0, sA); else loadA(0);
  storeW(sW);
  if constexpr (!ABF) storeA(sA);
  if constexpr (ABF) asm volatile("s_wait_asynccnt 0x0" ::: "memory");
  __syncthreads();

  // ---- pipelined K loop: 1 barrier per K-step ------------------------------
  for (int kk = 0; kk < Klim; kk += 32) {
    __bf16* cA = (kk & 32) ? sA1 : sA;
    __bf16* cW = (kk & 32) ? sW1 : sW;
    __bf16* nA = (kk & 32) ? sA : sA1;
    __bf16* nW = (kk & 32) ? sW : sW1;
    const bool more = (kk + 32) < Klim;     // uniform branch (EXEC stays full)

    if (more) {
      loadW(kk + 32);                        // global b128 loads, no wait yet
      if constexpr (ABF) asyncA(kk + 32, nA); else loadA(kk + 32);
    }
    compute(cA, cW);                         // WMMAs hide load/DMA latency
    if (more) {
      storeW(nW);                            // cvt + ds_store (waits loadcnt here)
      if constexpr (!ABF) storeA(nA);
      if constexpr (ABF) asm volatile("s_wait_asynccnt 0x0" ::: "memory");
      __syncthreads();
    }
  }
  __syncthreads();  // buffers may be recycled by the next mm_accum phase
}

// ---------------------------------------------------------------------------
// One LSTM layer; epilogue fuses gate math and writes leaky_relu(h) as bf16.
// ---------------------------------------------------------------------------
template <bool ABF>
__global__ void __launch_bounds__(NTHREADS)
lstm_layer_kernel(const void* __restrict__ X, int ldx, int Kx,
                  const float* __restrict__ H0, const float* __restrict__ C0,
                  const float* __restrict__ Wih, int ldw_ih,
                  const float* __restrict__ Whh,
                  const float* __restrict__ Bih, const float* __restrict__ Bhh,
                  __bf16* __restrict__ Xout)
{
  __shared__ __bf16 sA[2 * BM * LSTR];          // 20 KB (double buffered)
  __shared__ __bf16 sW[2 * 4 * BN * LSTR];      // 20 KB

  const int tid = threadIdx.x;
  const int m0 = blockIdx.x * BM;
  const int n0 = blockIdx.y * BN;

  v8f acc[4][2];
  const v8f zero = {0.f, 0.f, 0.f, 0.f, 0.f, 0.f, 0.f, 0.f};
#pragma unroll
  for (int g = 0; g < 4; ++g)
#pragma unroll
    for (int ct = 0; ct < 2; ++ct) acc[g][ct] = zero;

  mm_accum<4, ABF>(X, ldx, Kx, Wih, ldw_ih, HDIM, HDIM, m0, n0, tid, sA, sW, acc);
  mm_accum<4, false>(H0, HDIM, HDIM, Whh, HDIM, HDIM, HDIM, m0, n0, tid, sA, sW, acc);

  // C/D layout: VGPR v, lane L -> M = v + 8*(L>>4), N = L&15 (ISA 7.12.2).
  const int lane = tid & 31;
  const int wv   = tid >> 5;
#pragma unroll
  for (int ct = 0; ct < 2; ++ct) {
    const int nc = n0 + ct * 16 + (lane & 15);
    const float bI = Bih[nc]            + Bhh[nc];
    const float bF = Bih[HDIM + nc]     + Bhh[HDIM + nc];
    const float bG = Bih[2 * HDIM + nc] + Bhh[2 * HDIM + nc];
    const float bO = Bih[3 * HDIM + nc] + Bhh[3 * HDIM + nc];
#pragma unroll
    for (int v = 0; v < 8; ++v) {
      const int m = m0 + (wv << 4) + v + ((lane >> 4) << 3);
      const float iv = acc[0][ct][v] + bI;
      const float fv = acc[1][ct][v] + bF;
      const float gv = acc[2][ct][v] + bG;
      const float ov = acc[3][ct][v] + bO;
      const float c  = C0[(size_t)m * HDIM + nc];
      const float cn = fsig(fv) * c + fsig(iv) * ftanh_fast(gv);
      const float hn = fsig(ov) * ftanh_fast(cn);
      Xout[(size_t)m * HDIM + nc] = (__bf16)((hn > 0.0f) ? hn : 0.01f * hn);
    }
  }
}

// ---------------------------------------------------------------------------
// Dense: Y = leaky_relu(X @ W^T + b); X bf16, output type OT (bf16 or fp32).
// ---------------------------------------------------------------------------
template <typename OT>
__global__ void __launch_bounds__(NTHREADS)
dense_lrelu_kernel(const __bf16* __restrict__ X, const float* __restrict__ W,
                   const float* __restrict__ Bias, OT* __restrict__ Y,
                   int Nout, int ldo)
{
  __shared__ __bf16 sA[2 * BM * LSTR];          // 20 KB
  __shared__ __bf16 sW[2 * BN * LSTR];          // 5 KB

  const int tid = threadIdx.x;
  const int m0 = blockIdx.x * BM;
  const int n0 = blockIdx.y * BN;

  v8f acc[1][2];
  const v8f zero = {0.f, 0.f, 0.f, 0.f, 0.f, 0.f, 0.f, 0.f};
  acc[0][0] = zero;
  acc[0][1] = zero;

  mm_accum<1, true>(X, HDIM, HDIM, W, HDIM, 0, Nout, m0, n0, tid, sA, sW, acc);

  const int lane = tid & 31;
  const int wv   = tid >> 5;
#pragma unroll
  for (int ct = 0; ct < 2; ++ct) {
    const int nc = n0 + ct * 16 + (lane & 15);
    if (nc < Nout) {
      const float b = Bias[nc];
#pragma unroll
      for (int v = 0; v < 8; ++v) {
        const int m = m0 + (wv << 4) + v + ((lane >> 4) << 3);
        const float h = acc[0][ct][v] + b;
        Y[(size_t)m * ldo + nc] = (OT)((h > 0.0f) ? h : 0.01f * h);
      }
    }
  }
}

// Pad a (rows x ksrc) fp32 matrix to (rows x 32) with zeros.
__global__ void pad32_kernel(const float* __restrict__ src, float* __restrict__ dst,
                             int rows, int ksrc)
{
  const int i = blockIdx.x * NTHREADS + threadIdx.x;
  if (i < rows * 32) {
    const int r = i >> 5, c = i & 31;
    dst[i] = (c < ksrc) ? src[r * ksrc + c] : 0.0f;
  }
}

// ---------------------------------------------------------------------------
extern "C" void kernel_launch(void* const* d_in, const int* in_sizes, int n_in,
                              void* d_out, int out_size, void* d_ws, size_t ws_size,
                              hipStream_t stream) {
  (void)in_sizes; (void)n_in; (void)out_size; (void)ws_size;

  const int L = 10, B = 1024, H = 1024, I = 17, O = 16;
  const float* x    = (const float*)d_in[0];
  const float* h0   = (const float*)d_in[1];
  const float* c0   = (const float*)d_in[2];
  const float* wih0 = (const float*)d_in[3];
  const float* bih0 = (const float*)d_in[4];
  const float* wih  = (const float*)d_in[5];
  const float* bih  = (const float*)d_in[6];
  const float* whh  = (const float*)d_in[7];
  const float* bhh  = (const float*)d_in[8];
  const float* w4   = (const float*)d_in[9];
  const float* b4   = (const float*)d_in[10];
  const float* w5   = (const float*)d_in[11];
  const float* b5   = (const float*)d_in[12];
  const float* w6   = (const float*)d_in[13];
  const float* b6   = (const float*)d_in[14];

  // workspace: | w0pad f32 4096x32 | x0pad f32 1024x32 | bufA bf16 BxH | bufB bf16 BxH |
  float*  w0pad = (float*)d_ws;
  float*  x0pad = w0pad + 4096 * 32;
  __bf16* bufA  = (__bf16*)(x0pad + 1024 * 32);
  __bf16* bufB  = bufA + (size_t)B * H;

  const dim3 blk(NTHREADS);
  const dim3 grd(B / BM, H / BN);              // 8 x 32

  // Pre-pad the K=17 operands to K=32 (removes all K guards from hot loops).
  pad32_kernel<<<(4096 * 32) / NTHREADS, blk, 0, stream>>>(wih0, w0pad, 4096, I);
  pad32_kernel<<<(1024 * 32) / NTHREADS, blk, 0, stream>>>(x, x0pad, 1024, I);

  // Layer 0: padded fp32 A, K = 32.
  lstm_layer_kernel<false><<<grd, blk, 0, stream>>>(
      x0pad, 32, 32, h0, c0, w0pad, 32, whh, bih0, bhh, bufA);

  __bf16* cur = bufA;
  __bf16* nxt = bufB;
  for (int l = 1; l < L; ++l) {
    lstm_layer_kernel<true><<<grd, blk, 0, stream>>>(
        cur, H, H,
        h0 + (size_t)l * B * H, c0 + (size_t)l * B * H,
        wih + (size_t)(l - 1) * 4 * H * H, H,
        whh + (size_t)l * 4 * H * H,
        bih + (size_t)(l - 1) * 4 * H,
        bhh + (size_t)l * 4 * H,
        nxt);
    __bf16* t = cur; cur = nxt; nxt = t;
  }

  dense_lrelu_kernel<__bf16><<<grd, blk, 0, stream>>>(cur, w4, b4, nxt, H, H);
  { __bf16* t = cur; cur = nxt; nxt = t; }
  dense_lrelu_kernel<__bf16><<<grd, blk, 0, stream>>>(cur, w5, b5, nxt, H, H);
  { __bf16* t = cur; cur = nxt; nxt = t; }
  dense_lrelu_kernel<float><<<dim3(B / BM, 1), blk, 0, stream>>>(
      cur, w6, b6, (float*)d_out, O, O);
}
